// PSALMFusion_39694087750159
// MI455X (gfx1250) — matrix operand order repaired and checked by reading, not compiled, not run on hardware
//
#include <hip/hip_runtime.h>
#include <hip/hip_bf16.h>
#include <math.h>

// ---------------------------------------------------------------------------
// MI455X (gfx1250) implementation of the PSALM pyramid+linear-attention block.
// All 1x1 convolutions run as WMMA GEMMs (V_WMMA_F32_16X16X32_F16, fp32
// accumulate). Activations/weights are pre-packed into the exact CDNA5 WMMA
// per-lane VGPR order as f16 (one coalesced pass). The GEMM wave computes a
// 16(oc) x 32(n) tile: shared A fragment, two WMMAs per K-chunk, manually
// double-buffered so next-chunk b128 loads overlap current-chunk WMMAs.
// ---------------------------------------------------------------------------

typedef __attribute__((ext_vector_type(16))) _Float16 v16h;
typedef __attribute__((ext_vector_type(8)))  float    v8f;

#define EPS 1e-5f

// ===========================================================================
// Pack weights w[OC][IC] (f32) into WMMA A-matrix lane order, f16, K padded
// to multiple of 32.  Layout: pa[((ot*KC + c)*32 + lane)*16 + e]
// A 16x32 f16 tile: M = lane&15; elem e -> v=e>>1, h=e&1,
//   K = c*32 + ((v<4)? 2v : 16+2(v-4)) + 8*(lane>=16) + h
// ===========================================================================
__global__ void k_packA(const float* __restrict__ w, _Float16* __restrict__ pa,
                        int IC, int OC, int KC)
{
    const int total = (OC / 16) * KC * 32;
    const int idx = blockIdx.x * blockDim.x + threadIdx.x;
    if (idx >= total) return;
    const int lane = idx & 31;
    const int c    = (idx >> 5) % KC;
    const int ot   = idx / (32 * KC);
    const int lg = lane >> 4, lr = lane & 15;
    const float* __restrict__ wrow = w + (size_t)(ot * 16 + lr) * IC;

    v16h out;
#pragma unroll
    for (int e = 0; e < 16; ++e) {
        const int v = e >> 1, h = e & 1;
        const int ka = c * 32 + ((v < 4) ? (2 * v) : (16 + 2 * (v - 4))) + 8 * lg + h;
        out[e] = (_Float16)(ka < IC ? wrow[ka] : 0.0f);
    }
    *(v16h*)(pa + (size_t)idx * 16) = out;
}

// ===========================================================================
// Pack activations x[B][IC][N] (f32) into WMMA B-matrix lane order, f16,
// K padded to multiple of 32.  Layout: pb[(((b*NT+nt)*KC + c)*32 + lane)*16+e]
// B 32x16 f16 tile: N-col = lane&15; elem e -> K = c*32 + ((e<8)? e : 16+(e-8)) + 8*(lane>=16)
// Reads coalesce along n across lanes; each lane writes 32 contiguous bytes.
// ===========================================================================
__global__ void k_packB(const float* __restrict__ x, _Float16* __restrict__ pb,
                        int IC, int N, int KC)
{
    const int NT = N / 16;
    const long long total = (long long)16 * NT * KC * 32;
    long long idx = (long long)blockIdx.x * blockDim.x + threadIdx.x;
    if (idx >= total) return;
    const int lane = (int)(idx & 31);
    const int c    = (int)((idx >> 5) % KC);
    const int nt   = (int)((idx / (32LL * KC)) % NT);
    const int b    = (int)(idx / (32LL * KC * NT));
    const int lg = lane >> 4, lr = lane & 15;
    const float* __restrict__ xb = x + (size_t)b * IC * N + nt * 16 + lr;

    v16h out;
#pragma unroll
    for (int e = 0; e < 16; ++e) {
        const int k = c * 32 + ((e < 8) ? e : (16 + (e - 8))) + 8 * lg;
        out[e] = (_Float16)(k < IC ? xb[(size_t)k * N] : 0.0f);
    }
    *(v16h*)(pb + (size_t)idx * 16) = out;
}

// ===========================================================================
// GEMM on pre-packed tiles:  y[b,oc,n] = bias[oc] + sum_k w*x (+ residual)
// One wave computes 16(oc) x 32(n): one shared A fragment, two accumulators,
// double-buffered K-chunks. 4 waves/block.
// ===========================================================================
__global__ __launch_bounds__(128)
void k_gemm_packed(const _Float16* __restrict__ pa, const _Float16* __restrict__ pb,
                   const float* __restrict__ bias, const float* __restrict__ resid,
                   float* __restrict__ y, int KC, int OC, int N)
{
    const int lane = threadIdx.x & 31;
    const int wave = threadIdx.x >> 5;
    const int lg = lane >> 4, lr = lane & 15;
    const int NT = N / 16;

    const int nt0 = (blockIdx.x * 4 + wave) * 2;   // pair of adjacent n-tiles
    const int ot  = blockIdx.y;
    const int b   = blockIdx.z;

    const _Float16* __restrict__ ap  = pa + ((size_t)(ot * KC) * 32 + lane) * 16;
    const _Float16* __restrict__ bp0 = pb + (((size_t)(b * NT + nt0) * KC) * 32 + lane) * 16;
    const _Float16* __restrict__ bp1 = bp0 + (size_t)KC * 512;   // next n-tile

    v16h A  = *(const v16h*)ap;
    v16h B0 = *(const v16h*)bp0;
    v16h B1 = *(const v16h*)bp1;

    v8f acc0 = {}, acc1 = {};
    for (int c = 0; c < KC - 1; ++c) {
        ap += 512; bp0 += 512; bp1 += 512;
        const v16h An  = *(const v16h*)ap;     // next-chunk loads issue first,
        const v16h B0n = *(const v16h*)bp0;    // overlap with the two WMMAs
        const v16h B1n = *(const v16h*)bp1;
        acc0 = __builtin_amdgcn_wmma_f32_16x16x32_f16(
                   false, A, false, B0, (short)0, acc0, false, false);
        acc1 = __builtin_amdgcn_wmma_f32_16x16x32_f16(
                   false, A, false, B1, (short)0, acc1, false, false);
        A = An; B0 = B0n; B1 = B1n;
    }
    acc0 = __builtin_amdgcn_wmma_f32_16x16x32_f16(
               false, A, false, B0, (short)0, acc0, false, false);
    acc1 = __builtin_amdgcn_wmma_f32_16x16x32_f16(
               false, A, false, B1, (short)0, acc1, false, false);

    // D tiles: 8 f32 rows each, m = r + 8*lg, cols nt0*16+lr and (nt0+1)*16+lr
#pragma unroll
    for (int r = 0; r < 8; ++r) {
        const int oc = ot * 16 + r + 8 * lg;
        const float bv = bias[oc];
        const size_t off0 = ((size_t)b * OC + oc) * N + nt0 * 16 + lr;
        float v0 = acc0[r] + bv;
        float v1 = acc1[r] + bv;
        if (resid) { v0 += resid[off0]; v1 += resid[off0 + 16]; }
        y[off0]      = v0;
        y[off0 + 16] = v1;
    }
}

// ===========================================================================
// Generic depthwise / channel-expanding small conv.
//   ic = oc * Cin / Cout  (Cin==Cout: depthwise; Cout==2*Cin: bu grouped conv)
//   optional fused BN (given stats), SiLU, residual add.
// ===========================================================================
__global__ void k_dwconv(const float* __restrict__ x, const float* __restrict__ w,
                         const float* __restrict__ bias,
                         const float* __restrict__ bn_g, const float* __restrict__ bn_b,
                         const float* __restrict__ bn_rm, const float* __restrict__ bn_rv,
                         const float* __restrict__ resid, float* __restrict__ y,
                         int Bn, int Cin, int Cout, int Hin, int Win,
                         int Hout, int Wout, int ks, int stride, int pad, int do_silu)
{
    const long long total = (long long)Bn * Cout * Hout * Wout;
    long long idx = (long long)blockIdx.x * blockDim.x + threadIdx.x;
    if (idx >= total) return;
    const int ox = (int)(idx % Wout);
    const int oy = (int)((idx / Wout) % Hout);
    const int oc = (int)((idx / ((long long)Wout * Hout)) % Cout);
    const int b  = (int)(idx / ((long long)Wout * Hout * Cout));
    const int ic = (int)((long long)oc * Cin / Cout);

    const float* __restrict__ xp = x + ((size_t)b * Cin + ic) * Hin * Win;
    const float* __restrict__ wp = w + (size_t)oc * ks * ks;

    float acc = bias[oc];
    for (int kh = 0; kh < ks; ++kh) {
        const int iy = oy * stride - pad + kh;
        if (iy < 0 || iy >= Hin) continue;
        for (int kw = 0; kw < ks; ++kw) {
            const int ix = ox * stride - pad + kw;
            if (ix < 0 || ix >= Win) continue;
            acc += wp[kh * ks + kw] * xp[(size_t)iy * Win + ix];
        }
    }
    if (bn_g) {
        acc = (acc - bn_rm[oc]) * rsqrtf(bn_rv[oc] + EPS) * bn_g[oc] + bn_b[oc];
    }
    if (do_silu) acc = acc / (1.0f + __expf(-acc));
    if (resid) acc += resid[idx];
    y[idx] = acc;
}

// ===========================================================================
// Bilinear 2x upsample (half-pixel centers, edge clamp, == jax.image.resize)
// fused with residual add:  y = base + up2(coarse)
// ===========================================================================
__global__ void k_up2_add(const float* __restrict__ coarse, const float* __restrict__ base,
                          float* __restrict__ y, int Bn, int C, int Hc, int Wc)
{
    const int Ho = Hc * 2, Wo = Wc * 2;
    const long long total = (long long)Bn * C * Ho * Wo;
    long long idx = (long long)blockIdx.x * blockDim.x + threadIdx.x;
    if (idx >= total) return;
    const int ox = (int)(idx % Wo);
    const int oy = (int)((idx / Wo) % Ho);
    const int c  = (int)((idx / ((long long)Wo * Ho)) % C);
    const int b  = (int)(idx / ((long long)Wo * Ho * C));

    const float sy = (oy + 0.5f) * 0.5f - 0.5f;
    const float sx = (ox + 0.5f) * 0.5f - 0.5f;
    int y0i = (int)floorf(sy), x0i = (int)floorf(sx);
    const float fy = sy - y0i, fx = sx - x0i;
    const int y0 = min(max(y0i, 0), Hc - 1), y1 = min(max(y0i + 1, 0), Hc - 1);
    const int x0 = min(max(x0i, 0), Wc - 1), x1 = min(max(x0i + 1, 0), Wc - 1);

    const float* __restrict__ p = coarse + ((size_t)b * C + c) * Hc * Wc;
    const float v00 = p[(size_t)y0 * Wc + x0], v01 = p[(size_t)y0 * Wc + x1];
    const float v10 = p[(size_t)y1 * Wc + x0], v11 = p[(size_t)y1 * Wc + x1];
    const float v = v00 * (1 - fy) * (1 - fx) + v01 * (1 - fy) * fx +
                    v10 * fy * (1 - fx)       + v11 * fy * fx;
    y[idx] = base[idx] + v;
}

// ===========================================================================
// Row softmax (K over spatial axis), in-place. One 256-thread block per row.
// ===========================================================================
__global__ __launch_bounds__(256)
void k_softmax_rows(float* __restrict__ x, int n)
{
    float* __restrict__ p = x + (size_t)blockIdx.x * n;
    __shared__ float sd[256];
    const int t = threadIdx.x;

    float m = -INFINITY;
    for (int i = t; i < n; i += 256) m = fmaxf(m, p[i]);
    sd[t] = m; __syncthreads();
    for (int s = 128; s > 0; s >>= 1) {
        if (t < s) sd[t] = fmaxf(sd[t], sd[t + s]);
        __syncthreads();
    }
    m = sd[0]; __syncthreads();

    float sum = 0.0f;
    for (int i = t; i < n; i += 256) sum += __expf(p[i] - m);
    sd[t] = sum; __syncthreads();
    for (int s = 128; s > 0; s >>= 1) {
        if (t < s) sd[t] += sd[t + s];
        __syncthreads();
    }
    const float inv = 1.0f / sd[0];
    for (int i = t; i < n; i += 256) p[i] = __expf(p[i] - m) * inv;
}

// ===========================================================================
// Channel softmax (Q over kh channels within a head), in-place.
// ===========================================================================
__global__ void k_softmax_chan(float* __restrict__ q, int Bn, int kc, int kh, int N)
{
    const long long total = (long long)Bn * 4 * N;
    long long idx = (long long)blockIdx.x * blockDim.x + threadIdx.x;
    if (idx >= total) return;
    const int n = (int)(idx % N);
    const int h = (int)((idx / N) % 4);
    const int b = (int)(idx / ((long long)N * 4));
    float* __restrict__ p = q + ((size_t)b * kc + (size_t)h * kh) * N + n;

    float m = -INFINITY;
    for (int k = 0; k < kh; ++k) m = fmaxf(m, p[(size_t)k * N]);
    float sum = 0.0f;
    for (int k = 0; k < kh; ++k) sum += __expf(p[(size_t)k * N] - m);
    const float inv = 1.0f / sum;
    for (int k = 0; k < kh; ++k) p[(size_t)k * N] = __expf(p[(size_t)k * N] - m) * inv;
}

// ===========================================================================
// ctx[b,h,k,v] = sum_n K[b, h*kh+k, n] * V[b, h*vh+v, n]
// ===========================================================================
__global__ __launch_bounds__(256)
void k_ctx(const float* __restrict__ K, const float* __restrict__ V,
           float* __restrict__ ctx, int kc, int vc, int kh, int vh, int N)
{
    const int v = blockIdx.x % vh;
    const int k = (blockIdx.x / vh) % kh;
    const int h = (blockIdx.x / (vh * kh)) % 4;
    const int b = blockIdx.x / (vh * kh * 4);
    const float* __restrict__ Kp = K + ((size_t)b * kc + (size_t)h * kh + k) * N;
    const float* __restrict__ Vp = V + ((size_t)b * vc + (size_t)h * vh + v) * N;

    __shared__ float sd[256];
    const int t = threadIdx.x;
    float sum = 0.0f;
    for (int i = t; i < N; i += 256) sum += Kp[i] * Vp[i];
    sd[t] = sum; __syncthreads();
    for (int s = 128; s > 0; s >>= 1) {
        if (t < s) sd[t] += sd[t + s];
        __syncthreads();
    }
    if (t == 0) ctx[blockIdx.x] = sd[0];
}

// ===========================================================================
// Support gate: wgt[b,j] = sigmoid(LN(proto[b] . mw[j] + mb[j]))
// ===========================================================================
__global__ __launch_bounds__(128)
void k_proto_gate(const float* __restrict__ proto, const float* __restrict__ mw,
                  const float* __restrict__ mb, const float* __restrict__ g,
                  const float* __restrict__ bb, float* __restrict__ wgt, int C, int vc)
{
    const int b = blockIdx.x;
    const int j = threadIdx.x;
    __shared__ float sd[128];

    float mj = 0.0f;
    if (j < vc) {
        mj = mb[j];
        const float* __restrict__ pr = proto + (size_t)b * C;
        const float* __restrict__ wr = mw + (size_t)j * C;
        for (int c = 0; c < C; ++c) mj += pr[c] * wr[c];
    }
    sd[j] = (j < vc) ? mj : 0.0f; __syncthreads();
    for (int s = 64; s > 0; s >>= 1) { if (j < s) sd[j] += sd[j + s]; __syncthreads(); }
    const float mean = sd[0] / vc; __syncthreads();
    const float d = (j < vc) ? (mj - mean) : 0.0f;
    sd[j] = d * d; __syncthreads();
    for (int s = 64; s > 0; s >>= 1) { if (j < s) sd[j] += sd[j + s]; __syncthreads(); }
    const float var = sd[0] / vc;
    if (j < vc) {
        const float z = d * rsqrtf(var + EPS) * g[j] + bb[j];
        wgt[(size_t)b * vc + j] = 1.0f / (1.0f + __expf(-z));
    }
}

// ===========================================================================
// att[b,j,n] = wgt[b,j] * sum_k ctx[b, h, k, j%vh] * Q[b, h*kh+k, n]
// ===========================================================================
__global__ void k_attn_out(const float* __restrict__ ctx, const float* __restrict__ Q,
                           const float* __restrict__ wgt, float* __restrict__ att,
                           int Bn, int kc, int vc, int kh, int vh, int N)
{
    const long long total = (long long)Bn * vc * N;
    long long idx = (long long)blockIdx.x * blockDim.x + threadIdx.x;
    if (idx >= total) return;
    const int n = (int)(idx % N);
    const int j = (int)((idx / N) % vc);
    const int b = (int)(idx / ((long long)N * vc));
    const int h = j / vh, v = j % vh;

    const float* __restrict__ cp = ctx + (((size_t)b * 4 + h) * kh) * vh + v;
    const float* __restrict__ qp = Q + ((size_t)b * kc + (size_t)h * kh) * N + n;
    float sum = 0.0f;
    for (int k = 0; k < kh; ++k) sum += cp[(size_t)k * vh] * qp[(size_t)k * N];
    att[idx] = sum * wgt[(size_t)b * vc + j];
}

// ===========================================================================
// Host orchestration
// ===========================================================================
static inline void launch_dw(hipStream_t s, const float* x, const float* w, const float* b,
                             const float* g, const float* bb, const float* rm, const float* rv,
                             const float* resid, float* y, int Bn, int Cin, int Cout,
                             int Hin, int Hout, int ks, int stride, int pad, int silu)
{
    long long total = (long long)Bn * Cout * Hout * Hout;
    int blocks = (int)((total + 255) / 256);
    k_dwconv<<<blocks, 256, 0, s>>>(x, w, b, g, bb, rm, rv, resid, y,
                                    Bn, Cin, Cout, Hin, Hin, Hout, Hout, ks, stride, pad, silu);
}

static inline void launch_gemm(hipStream_t s, _Float16* pa, _Float16* pb,
                               const float* x, const float* w, const float* b,
                               const float* resid, float* y, int Bimg, int IC, int OC, int N)
{
    const int KC = (IC + 31) / 32;
    const int NT = N / 16;
    {   // pack weights into A-tile order (tiny)
        int total = (OC / 16) * KC * 32;
        k_packA<<<(total + 255) / 256, 256, 0, s>>>(w, pa, IC, OC, KC);
    }
    {   // pack activations into B-tile order (one coalesced f32->f16 pass)
        long long total = (long long)Bimg * NT * KC * 32;
        k_packB<<<(int)((total + 255) / 256), 256, 0, s>>>(x, pb, IC, N, KC);
    }
    dim3 grid(NT / 8, OC / 16, Bimg);   // each wave: 2 n-tiles; 4 waves/block
    k_gemm_packed<<<grid, 128, 0, s>>>(pa, pb, b, resid, y, KC, OC, N);
}

extern "C" void kernel_launch(void* const* d_in, const int* in_sizes, int n_in,
                              void* d_out, int out_size, void* d_ws, size_t ws_size,
                              hipStream_t stream)
{
    (void)in_sizes; (void)n_in; (void)out_size; (void)ws_size;
    const int B = 16;
    const int CH[4]   = {32, 64, 128, 256};
    const int HW[4]   = {128, 64, 32, 16};
    const int OUTC[4] = {128, 256, 512, 512};

    // ---- input mapping (setup_inputs dict order, params flattened DFS) ----
    const float* q[4] = {(const float*)d_in[0], (const float*)d_in[2],
                         (const float*)d_in[4], (const float*)d_in[6]};
    const float* sp[4] = {(const float*)d_in[1], (const float*)d_in[3],
                          (const float*)d_in[5], (const float*)d_in[7]};
    auto P = [&](int i) { return (const float*)d_in[i]; };
    // pyramid: td_w 8-10, td_b 11-13, bu_w 14-16, bu_b 17-19, fu_w 20-23,
    //          fu_b 24-27, bn_g 28-31, bn_b 32-35, bn_rm 36-39, bn_rv 40-43
    // attn[i]: base 44 + 24*i, entries in dict order.

    // ---- workspace partition (floats) ----
    float* W = (float*)d_ws;
    size_t cur = 0;
    auto alloc = [&](size_t nf) { float* p = W + cur; cur += nf; return p; };
    const size_t featN[4] = {(size_t)B * 32 * 16384, (size_t)B * 64 * 4096,
                             (size_t)B * 128 * 1024, (size_t)B * 256 * 256};
    float* tmp = alloc(featN[0] / 4);                 // td conv scratch
    float* td0 = alloc(featN[0]);
    float* td1 = alloc(featN[1]);
    float* td2 = alloc(featN[2]);
    float* bu1 = alloc(featN[1]);
    float* bu2 = alloc(featN[2]);
    float* bu3 = alloc(featN[3]);
    float* E[4];
    for (int i = 0; i < 4; ++i) E[i] = alloc(featN[i]);
    float* T1 = alloc(featN[0]);
    float* P1 = alloc(featN[0]);
    float* P2 = alloc(featN[0]);
    float* Qb = alloc(featN[0] / 2);
    float* Kb = alloc(featN[0] / 2);
    float* Vb = alloc(featN[0] / 2);
    float* CTX = alloc((size_t)B * 4 * 32 * 32);
    float* WGT = alloc((size_t)B * 128);
    _Float16* PA = (_Float16*)alloc(64 * 1024);           // packed weights (f16)
    _Float16* PB = (_Float16*)alloc(featN[0] / 2 + 64);   // packed activations (f16)

    // =======================  Stage 1: top-down  ==========================
    launch_gemm(stream, PA, PB, q[3], P(10), P(13), nullptr, tmp, B, 256, 128, 256);
    k_up2_add<<<(int)((featN[2] + 255) / 256), 256, 0, stream>>>(tmp, q[2], td2, B, 128, 16, 16);
    launch_gemm(stream, PA, PB, td2, P(9), P(12), nullptr, tmp, B, 128, 64, 1024);
    k_up2_add<<<(int)((featN[1] + 255) / 256), 256, 0, stream>>>(tmp, q[1], td1, B, 64, 32, 32);
    launch_gemm(stream, PA, PB, td1, P(8), P(11), nullptr, tmp, B, 64, 32, 4096);
    k_up2_add<<<(int)((featN[0] + 255) / 256), 256, 0, stream>>>(tmp, q[0], td0, B, 32, 64, 64);

    // =======================  Stage 1: bottom-up  =========================
    const float* bu0 = td0;
    launch_dw(stream, bu0, P(14), P(17), nullptr, nullptr, nullptr, nullptr, td1, bu1,
              B, 32, 64, 128, 64, 3, 2, 1, 0);
    launch_dw(stream, bu1, P(15), P(18), nullptr, nullptr, nullptr, nullptr, td2, bu2,
              B, 64, 128, 64, 32, 3, 2, 1, 0);
    launch_dw(stream, bu2, P(16), P(19), nullptr, nullptr, nullptr, nullptr, q[3], bu3,
              B, 128, 256, 32, 16, 3, 2, 1, 0);

    // ===========  Stage 1: fuse conv + BN + SiLU + residual  ==============
    const float* bu[4] = {bu0, bu1, bu2, bu3};
    for (int i = 0; i < 4; ++i)
        launch_dw(stream, bu[i], P(20 + i), P(24 + i), P(28 + i), P(32 + i), P(36 + i),
                  P(40 + i), q[i], E[i], B, CH[i], CH[i], HW[i], HW[i], 3, 1, 1, 1);

    // =============  Stage 2+3: per-scale linear attention  ================
    size_t out_off = 0;
    for (int i = 0; i < 4; ++i) {
        const int C = CH[i], Hh = HW[i], N = Hh * Hh;
        const int kc = C / 2, vc = C / 2, kh = kc / 4, vh = vc / 4;
        const int A = 44 + 24 * i;

        // short-long depthwise preprocessing for Q and K paths
        launch_dw(stream, E[i], P(A + 0), P(A + 1), nullptr, nullptr, nullptr, nullptr,
                  nullptr, T1, B, C, C, Hh, Hh, 3, 1, 1, 1);
        launch_dw(stream, T1, P(A + 2), P(A + 3), nullptr, nullptr, nullptr, nullptr,
                  nullptr, P1, B, C, C, Hh, Hh, 7, 1, 3, 0);
        launch_dw(stream, E[i], P(A + 4), P(A + 5), nullptr, nullptr, nullptr, nullptr,
                  nullptr, T1, B, C, C, Hh, Hh, 3, 1, 1, 1);
        launch_dw(stream, T1, P(A + 6), P(A + 7), nullptr, nullptr, nullptr, nullptr,
                  nullptr, P2, B, C, C, Hh, Hh, 7, 1, 3, 0);

        // Q/K/V 1x1 projections on WMMA
        launch_gemm(stream, PA, PB, P1, P(A + 8),  P(A + 9),  nullptr, Qb, B, C, kc, N);
        launch_gemm(stream, PA, PB, P2, P(A + 10), P(A + 11), nullptr, Kb, B, C, kc, N);
        launch_gemm(stream, PA, PB, E[i], P(A + 12), P(A + 13), nullptr, Vb, B, C, vc, N);

        // softmaxes
        k_softmax_rows<<<B * kc, 256, 0, stream>>>(Kb, N);
        {
            long long tot = (long long)B * 4 * N;
            k_softmax_chan<<<(int)((tot + 255) / 256), 256, 0, stream>>>(Qb, B, kc, kh, N);
        }

        // ctx = K V^T   and the support gate
        k_ctx<<<B * 4 * kh * vh, 256, 0, stream>>>(Kb, Vb, CTX, kc, vc, kh, vh, N);
        k_proto_gate<<<B, 128, 0, stream>>>(sp[i], P(A + 14), P(A + 15), P(A + 16),
                                            P(A + 17), WGT, C, vc);

        // att = (ctx^T Q) * gate      (into T1, free again)
        {
            long long tot = (long long)B * vc * N;
            k_attn_out<<<(int)((tot + 255) / 256), 256, 0, stream>>>(
                CTX, Qb, WGT, T1, B, kc, vc, kh, vh, N);
        }

        // fused = feats + rw(att)   (residual fused into the WMMA epilogue)
        launch_gemm(stream, PA, PB, T1, P(A + 18), P(A + 19), q[i], P1, B, vc, C, N);
        // r = SiLU(dw3(fused))
        launch_dw(stream, P1, P(A + 20), P(A + 21), nullptr, nullptr, nullptr, nullptr,
                  nullptr, P2, B, C, C, Hh, Hh, 3, 1, 1, 1);
        // out_i = ow(r)  -> final output slab
        launch_gemm(stream, PA, PB, P2, P(A + 22), P(A + 23), nullptr,
                    (float*)d_out + out_off, B, C, OUTC[i], N);
        out_off += (size_t)B * OUTC[i] * N;
    }
}
